// SA_Layer_88184268522190
// MI455X (gfx1250) — matrix operand order repaired and checked by reading, not compiled, not run on hardware
//
#include <hip/hip_runtime.h>
#include <hip/hip_bf16.h>

typedef unsigned int u32;
typedef unsigned short u16;
typedef unsigned long long u64;
typedef __attribute__((ext_vector_type(4)))  u32    u32x4;
typedef __attribute__((ext_vector_type(4)))  int    i32x4;
typedef __attribute__((ext_vector_type(8)))  int    i32x8;
typedef __attribute__((ext_vector_type(16))) __bf16 v16bf;
typedef __attribute__((ext_vector_type(8)))  float  v8f;

typedef __attribute__((address_space(1))) i32x4 gv4i;   // global int4
typedef __attribute__((address_space(3))) i32x4 lv4i;   // LDS int4

union Frag  { u32x4 q[2]; v16bf v; };
union Pack8 { u32x4 q; u16 s[8]; };

__device__ __forceinline__ u16 f2bf(float f) {
  u32 u = __float_as_uint(f);
  u32 r = (u + 0x7FFFu + ((u >> 16) & 1u)) >> 16;   // round-to-nearest-even
  return (u16)r;
}
__device__ __forceinline__ float bf2f(u16 h) {
  return __uint_as_float(((u32)h) << 16);
}

__device__ __forceinline__ v8f wmma_bf16(v16bf a, v16bf b, v8f c) {
  return __builtin_amdgcn_wmma_f32_16x16x32_bf16(false, a, false, b, (short)0, c, false, false);
}

// ---------------- CDNA5 async / TDM wrappers ---------------------------------------------
__device__ __forceinline__ void wait_tensor0() {
#if __has_builtin(__builtin_amdgcn_s_wait_tensorcnt)
  __builtin_amdgcn_s_wait_tensorcnt(0);
#else
  asm volatile("s_wait_tensorcnt 0x0" ::: "memory");
#endif
}

__device__ __forceinline__ void wait_async0() {
#if __has_builtin(__builtin_amdgcn_s_wait_asynccnt)
  __builtin_amdgcn_s_wait_asynccnt(0);
#else
  asm volatile("s_wait_asynccnt 0x0" ::: "memory");
#endif
}

// One lane-cooperative async 16B copy global -> LDS (GLOBAL_LOAD_ASYNC_TO_LDS_B128).
// clang-23 signature: (int4 AS1* src, int4 AS3* dst, imm offset, imm cpol)
__device__ __forceinline__ void async_ld_b128(const void* g, void* l) {
#if __has_builtin(__builtin_amdgcn_global_load_async_to_lds_b128)
  __builtin_amdgcn_global_load_async_to_lds_b128(
      (gv4i*)(uintptr_t)g, (lv4i*)(uintptr_t)l, 0, 0);
#else
  u32 lo = (u32)(uintptr_t)l;
  u64 ga = (u64)(uintptr_t)g;
  asm volatile("global_load_async_to_lds_b128 %0, %1, off" :: "v"(lo), "v"(ga) : "memory");
#endif
}

// TDM: DMA a 2-D tile of bf16 (rows x tile_w, row pitch stride0 elems) global -> LDS.
// D# per ISA ch.8: group0 {count=1, lds_addr, global_addr, type=2}; group1
// {data_size=1(2B), tensor_dim0/1, tile_dim0/1, tensor_dim0_stride}.
// clang-23 builtin arity: (uint32x4, int32x8, int32x4, int32x4, int32x8, i32 cpol)
__device__ __forceinline__ void tdm_load_tile_bf16(u32 lds_off, const void* gaddr,
                                                   u32 tile_w, u32 rows, u32 stride0) {
  u64 ga = (u64)(uintptr_t)gaddr;
  u32x4 g0;
  g0.x = 1u;                                            // count=1, user descriptor
  g0.y = lds_off;                                       // LDS byte address
  g0.z = (u32)ga;                                       // global_addr[31:0]
  g0.w = (u32)((ga >> 32) & 0x1FFFFFFu) | (2u << 30);   // global_addr[56:32] | type=2
  i32x8 g1;
  g1[0] = (int)(1u << 16);                              // wg_mask=0, data_size=1 (2 bytes)
  g1[1] = (int)((stride0 & 0xFFFFu) << 16);             // tensor_dim0[15:0] @bits 63:48
  g1[2] = (int)(((stride0 >> 16) & 0xFFFFu) | ((rows & 0xFFFFu) << 16)); // dim0 hi | dim1 lo
  g1[3] = (int)((tile_w & 0xFFFFu) << 16);              // tile_dim0 @bits 127:112
  g1[4] = (int)(rows & 0xFFFFu);                        // tile_dim1 (tile_dim2=0)
  g1[5] = (int)stride0;                                 // tensor_dim0_stride[31:0]
  g1[6] = 0;                                            // stride0 hi | stride1 lo
  g1[7] = 0;
#if __has_builtin(__builtin_amdgcn_tensor_load_to_lds)
  i32x4 z4 = {0, 0, 0, 0};
  i32x8 z8 = {0, 0, 0, 0, 0, 0, 0, 0};
  __builtin_amdgcn_tensor_load_to_lds(g0, g1, z4, z4, z8, 0);
#else
  asm volatile("tensor_load_to_lds %0, %1" :: "s"(g0), "s"(g1) : "memory");
#endif
}

// ---------------- WMMA fragment loads (contiguous k per lane half, same order in A and B) --------------------------
// A fragment: 16(M) x 32(K); lane l holds row (l&15); elems 0..15 = K k0+16*(l>>4)+s (contiguous).
__device__ __forceinline__ v16bf load_fragA(const u16* A, int lda, int r0, int k0, int lane) {
  const u16* p = A + (size_t)(r0 + (lane & 15)) * lda + k0 + ((lane >> 4) << 4);
  Frag f;
  f.q[0] = *(const u32x4*)(p);
  f.q[1] = *(const u32x4*)(p + 8);
  return f.v;
}
// B fragment: 32(K) x 16(N); column n = row (n0+l&15) of row-major [N,K]; 16 contiguous K.
__device__ __forceinline__ v16bf load_fragB(const u16* W, int ldw, int n0, int k0, int lane) {
  const u16* p = W + (size_t)(n0 + (lane & 15)) * ldw + k0 + ((lane >> 4) << 4);
  Frag f;
  f.q[0] = *(const u32x4*)(p);
  f.q[1] = *(const u32x4*)(p + 8);
  return f.v;
}

// ---------------- GEMM: out[M,Nout] = A[M,K] @ W[Nout,K]^T + bias (bf16 WMMA) -----------
// Block = 8 waves, tile 128(M) x 64(N). The 64x512 weight tile for the block is DMA'd
// into LDS once by the TDM; A streams from global; optional fused BN statistics.
__global__ __launch_bounds__(256) void gemm_nt_bf16(
    const u16* __restrict__ A, const u16* __restrict__ W,
    const float* __restrict__ bias, u16* __restrict__ out,
    int M, int Nout, int K, float* __restrict__ stats)
{
  __shared__ u16 Bs[64 * 512];                    // 64 KB weight tile (K = 512 here)

  const int lane  = threadIdx.x & 31;
  const int wave  = threadIdx.x >> 5;
  const int r0    = blockIdx.x * 128 + wave * 16;
  const int nbase = blockIdx.y * 64;

  if (wave == 0) {                                // one TDM descriptor per block
    tdm_load_tile_bf16((u32)(uintptr_t)Bs, W + (size_t)nbase * K, (u32)K, 64u, (u32)K);
    wait_tensor0();
  }
  __syncthreads();

  v8f acc[4];
  #pragma unroll
  for (int t = 0; t < 4; ++t) acc[t] = (v8f){0.f,0.f,0.f,0.f,0.f,0.f,0.f,0.f};

  for (int k0 = 0; k0 < K; k0 += 32) {
    v16bf a = load_fragA(A, K, r0, k0, lane);     // global (streamed, pipelined)
    #pragma unroll
    for (int t = 0; t < 4; ++t) {
      v16bf b = load_fragB(Bs, K, t * 16, k0, lane);  // LDS
      acc[t] = wmma_bf16(a, b, acc[t]);
    }
  }

  const int msub = (lane >> 4) << 3;              // C/D: row = vgpr + 8*(lane>=16)
  const int col0 = lane & 15;
  #pragma unroll
  for (int t = 0; t < 4; ++t) {
    int col = nbase + t * 16 + col0;
    float bv = bias[col];
    float s = 0.f, s2 = 0.f;
    #pragma unroll
    for (int vv = 0; vv < 8; ++vv) {
      int row = r0 + msub + vv;
      float val = acc[t][vv] + bv;
      out[(size_t)row * Nout + col] = f2bf(val);
      s += val; s2 += val * val;
    }
    if (stats) {
      s  += __shfl_xor(s, 16, 32);                // wave32 half-fold
      s2 += __shfl_xor(s2, 16, 32);
      if (lane < 16) {
        atomicAdd(&stats[col], s);
        atomicAdd(&stats[512 + col], s2);
      }
    }
  }
}

// ---------------- Per-crystal attention + x_trans ----------------------------------------
// LDS: Vt[512][72] bf16 | qks[64][256] bf16 (async-staged) | Vraw[64][512] bf16 (TDM-staged,
// region reused for E/colsum/att after the transpose barrier).
#define OFF_VT   0
#define OFF_QKS  (512*72*2)                 // 73728
#define OFF_VRAW (OFF_QKS + 64*256*2)       // 106496
#define ATT_SMEM (OFF_VRAW + 64*512*2)      // 172032

__global__ __launch_bounds__(256) void attn_xtrans(
    const u16* __restrict__ qk, const u16* __restrict__ v,
    const float* __restrict__ atom, u16* __restrict__ xt)
{
  extern __shared__ char smem[];
  u16*   Vt   = (u16*)(smem + OFF_VT);      // [512][72] transposed V
  u16*   qks  = (u16*)(smem + OFF_QKS);     // [64][256]
  u16*   Vraw = (u16*)(smem + OFF_VRAW);    // [64][512]
  float* E    = (float*)(smem + OFF_VRAW);  // [64][68]   (aliases Vraw, used after barrier)
  float* csum = E + 64 * 68;                // [64]
  u16*   attb = (u16*)(csum + 64);          // [64][64]

  const int tid  = threadIdx.x;
  const int lane = tid & 31;
  const int wave = tid >> 5;
  const int base = blockIdx.x * 64;

  // Phase 0: TDM the V tile (64x512 bf16, contiguous rows) and async-stage the qk tile.
  if (wave == 0)
    tdm_load_tile_bf16((u32)(uintptr_t)Vraw, v + (size_t)base * 512, 512u, 64u, 512u);
  #pragma unroll
  for (int it8 = 0; it8 < 8; ++it8) {              // 2048 x 16B chunks / 256 threads
    int g = tid + it8 * 256;
    async_ld_b128(qk + (size_t)base * 256 + g * 8, qks + g * 8);
  }
  wait_async0();
  if (wave == 0) wait_tensor0();
  __syncthreads();

  // Phase 1: transpose Vraw -> Vt (LDS -> LDS)
  for (int g = tid; g < 64 * 64; g += 256) {
    int j  = g >> 6;
    int c0 = (g & 63) << 3;
    Pack8 pk;
    pk.q = *(const u32x4*)(Vraw + j * 512 + c0);
    #pragma unroll
    for (int e = 0; e < 8; ++e) Vt[(c0 + e) * 72 + j] = pk.s[e];
  }
  __syncthreads();                                 // Vraw dead; E region live from here

  // Phase 2: energy = QK QK^T / 16 (4x4 tile grid, 2 tiles per wave), operands from LDS
  {
    const int it = wave >> 1;
    const int jb = (wave & 1) * 2;
    v8f c0 = (v8f){0.f,0.f,0.f,0.f,0.f,0.f,0.f,0.f}, c1 = c0;
    for (int k0 = 0; k0 < 256; k0 += 32) {
      v16bf a  = load_fragA(qks, 256, it * 16, k0, lane);
      v16bf b0 = load_fragB(qks, 256, jb * 16, k0, lane);
      v16bf b1 = load_fragB(qks, 256, (jb + 1) * 16, k0, lane);
      c0 = wmma_bf16(a, b0, c0);
      c1 = wmma_bf16(a, b1, c1);
    }
    const int m0 = it * 16 + ((lane >> 4) << 3);
    const int cc = lane & 15;
    #pragma unroll
    for (int vv = 0; vv < 8; ++vv) {
      E[(m0 + vv) * 68 + jb * 16 + cc]       = c0[vv] * 0.0625f;   // / sqrt(256)
      E[(m0 + vv) * 68 + (jb + 1) * 16 + cc] = c1[vv] * 0.0625f;
    }
  }
  __syncthreads();

  // Phase 3: softmax over query axis (one thread per key column)
  if (tid < 64) {
    int j = tid;
    float m = -3.0e38f;
    for (int i = 0; i < 64; ++i) m = fmaxf(m, E[i * 68 + j]);
    float s = 0.f;
    for (int i = 0; i < 64; ++i) { float e = __expf(E[i * 68 + j] - m); E[i * 68 + j] = e; s += e; }
    csum[j] = s;
  }
  __syncthreads();

  // Phase 4: L1-normalize over key axis; emit bf16 attention
  if (tid < 64) {
    int i = tid;
    float rs = 0.f;
    for (int j = 0; j < 64; ++j) rs += E[i * 68 + j] / csum[j];
    float inv = 1.f / (1e-9f + rs);
    for (int j = 0; j < 64; ++j) attb[i * 64 + j] = f2bf(E[i * 68 + j] / csum[j] * inv);
  }
  __syncthreads();

  // Phase 5: x_r = att @ V ; xt = atom - x_r (wave: 16 column-tiles, K=64 = 2 wmma steps)
  {
    const int it = wave >> 1;
    const int cg = wave & 1;
    v16bf a0 = load_fragA(attb, 64, it * 16, 0,  lane);
    v16bf a1 = load_fragA(attb, 64, it * 16, 32, lane);
    const int msub = (lane >> 4) << 3;
    const int cl   = lane & 15;
    const int kh   = (lane >> 4) << 4;
    for (int nt = 0; nt < 16; ++nt) {
      int n0 = cg * 256 + nt * 16;
      const u16* p0 = Vt + (size_t)(n0 + cl) * 72 + kh;
      Frag fb0, fb1;
      fb0.q[0] = *(const u32x4*)(p0);
      fb0.q[1] = *(const u32x4*)(p0 + 8);
      fb1.q[0] = *(const u32x4*)(p0 + 32);
      fb1.q[1] = *(const u32x4*)(p0 + 40);
      v8f c = (v8f){0.f,0.f,0.f,0.f,0.f,0.f,0.f,0.f};
      c = wmma_bf16(a0, fb0.v, c);
      c = wmma_bf16(a1, fb1.v, c);
      int col = n0 + cl;
      #pragma unroll
      for (int vv = 0; vv < 8; ++vv) {
        int grow = base + it * 16 + msub + vv;
        float xv = atom[(size_t)grow * 512 + col] - c[vv];
        xt[(size_t)grow * 512 + col] = f2bf(xv);
      }
    }
  }
}

// ---------------- small helper kernels ---------------------------------------------------
__global__ void cvt_f32_bf16(const float* __restrict__ in, u16* __restrict__ out, long n) {
  long i = (long)blockIdx.x * blockDim.x + threadIdx.x;
  long stride = (long)gridDim.x * blockDim.x;
  for (; i < n; i += stride) out[i] = f2bf(in[i]);
}

__global__ void zero_f32(float* p, int n) {
  int i = blockIdx.x * blockDim.x + threadIdx.x;
  if (i < n) p[i] = 0.f;
}

__global__ void bn_finalize(float* stats, const float* __restrict__ gamma,
                            const float* __restrict__ beta, float invN) {
  int c = threadIdx.x;
  float s = stats[c], sq = stats[512 + c];
  float mean = s * invN;
  float var  = fmaxf(sq * invN - mean * mean, 0.f);
  float sc   = gamma[c] * rsqrtf(var + 1e-5f);
  stats[c]       = sc;
  stats[512 + c] = beta[c] - mean * sc;
}

__global__ void final_residual(const float* __restrict__ a, const u16* __restrict__ t,
                               const float* __restrict__ stats, float* __restrict__ out, long n) {
  long i = (long)blockIdx.x * blockDim.x + threadIdx.x;
  long stride = (long)gridDim.x * blockDim.x;
  for (; i < n; i += stride) {
    int c = (int)(i & 511);
    float val = bf2f(t[i]) * stats[c] + stats[512 + c];
    out[i] = a[i] + fmaxf(val, 0.f);
  }
}

// ---------------- launch ------------------------------------------------------------------
extern "C" void kernel_launch(void* const* d_in, const int* in_sizes, int n_in,
                              void* d_out, int out_size, void* d_ws, size_t ws_size,
                              hipStream_t stream)
{
  const float* atom  = (const float*)d_in[0];
  const float* Wqk   = (const float*)d_in[1];
  const float* bqk   = (const float*)d_in[2];
  const float* Wv    = (const float*)d_in[3];
  const float* bv    = (const float*)d_in[4];
  const float* Wt    = (const float*)d_in[5];
  const float* bt    = (const float*)d_in[6];
  const float* gamma = (const float*)d_in[7];
  const float* beta  = (const float*)d_in[8];
  float* out = (float*)d_out;
  (void)in_sizes; (void)n_in; (void)out_size; (void)ws_size;

  const int N = 65536, C = 512, DK = 256, Bc = 1024;
  const long NC = (long)N * C;

  char* ws = (char*)d_ws;
  u16* wqk_bf = (u16*)(ws);                        // 256*512 bf16
  u16* wv_bf  = wqk_bf + DK * C;                   // 512*512 bf16
  u16* wt_bf  = wv_bf + C * C;                     // 512*512 bf16
  u16* a_bf   = (u16*)(ws + (2l  << 20));          // N*C bf16; reused as xt_bf16
  u16* qk_bf  = (u16*)(ws + (66l << 20));          // N*DK bf16
  u16* v_bf   = (u16*)(ws + (98l << 20));          // N*C bf16; reused as t_bf16
  float* stats = (float*)(ws + (162l << 20));      // [1024] f32
  u16* xt_bf = a_bf;
  u16* t_bf  = v_bf;

  // 1) fp32 -> bf16 staging
  cvt_f32_bf16<<<4096, 256, 0, stream>>>(atom, a_bf, NC);
  cvt_f32_bf16<<<512,  256, 0, stream>>>(Wqk, wqk_bf, (long)DK * C);
  cvt_f32_bf16<<<1024, 256, 0, stream>>>(Wv,  wv_bf,  (long)C * C);
  cvt_f32_bf16<<<1024, 256, 0, stream>>>(Wt,  wt_bf,  (long)C * C);

  // 2) projections (TDM-staged weights, bf16 WMMA, f32 accumulate)
  gemm_nt_bf16<<<dim3(N/128, DK/64), 256, 0, stream>>>(a_bf, wqk_bf, bqk, qk_bf, N, DK, C, nullptr);
  gemm_nt_bf16<<<dim3(N/128, C/64),  256, 0, stream>>>(a_bf, wv_bf,  bv,  v_bf,  N, C,  C, nullptr);

  // 3) per-crystal attention, fused x_trans = atom - att@V
  attn_xtrans<<<Bc, 256, ATT_SMEM, stream>>>(qk_bf, v_bf, atom, xt_bf);

  // 4) t = xt Wt^T + bt, fused BatchNorm statistics
  zero_f32<<<4, 256, 0, stream>>>(stats, 1024);
  gemm_nt_bf16<<<dim3(N/128, C/64), 256, 0, stream>>>(xt_bf, wt_bf, bt, t_bf, N, C, C, stats);

  // 5) BN finalize + fused BN/ReLU/residual
  bn_finalize<<<1, 512, 0, stream>>>(stats, gamma, beta, 1.0f / N);
  final_residual<<<8192, 256, 0, stream>>>(atom, t_bf, stats, out, NC);
}
